// CaptureCrossAttentionProcessor_66254165508885
// MI455X (gfx1250) — matrix operand order repaired and checked by reading, not compiled, not run on hardware
//
#include <hip/hip_runtime.h>
#include <hip/hip_bf16.h>

typedef __attribute__((ext_vector_type(16))) __bf16 v16bf;
typedef __attribute__((ext_vector_type(8)))  float  v8f;

#define WMMA_BF16(a, b, c) \
  __builtin_amdgcn_wmma_f32_16x16x32_bf16(false, (a), false, (b), (short)0, (c), false, false)

static constexpr int B_  = 8;
static constexpr int C_  = 640;   // channels = HEADS * DH
static constexpr int N_  = 4096;  // H*W tokens
static constexpr int L_  = 77;    // encoder tokens
static constexpr int Lp  = 96;    // L padded to 3*32 for K=32 WMMA steps
static constexpr int CD_ = 768;   // encoder dim
static constexpr int NH  = 8;     // heads
static constexpr int DH  = 80;    // head dim (padded to 96 at fragment load)

// ---------------------------------------------------------------------------
// Fragment loaders (CDNA5 WMMA bf16 16x16x32 layouts, CDNA5 ISA 7.12.2)
// A (16xK): lane -> M = lane&15; elems: k = kf0 + (lane>>4)*8 + (e&7) + (e>>3)*16
// B (Kx16): loaded from B^T stored row-major (n,k): lane -> N = lane&15;
//           elems: k = kf0 + (lane>>4)*8 + (e&7) + (e>>3)*16  (same as A)
// C/D (16x16 f32): lane -> N = lane&15; vgpr r -> M = r + 8*(lane>>4)
// ---------------------------------------------------------------------------
__device__ __forceinline__ v16bf load_a_bf16(const __bf16* base, int ld, int kf0, int kmax) {
  const int lane = threadIdx.x & 31;
  const __bf16* p = base + (size_t)(lane & 15) * ld;
  const int kb = kf0 + ((lane >> 4) << 3);
  v16bf a;
#pragma unroll
  for (int e = 0; e < 16; ++e) {
    int k = kb + (e & 7) + ((e >> 3) << 4);
    a[e] = (k < kmax) ? p[k] : (__bf16)0.0f;
  }
  return a;
}

__device__ __forceinline__ v16bf load_a_f32(const float* base, int ld, int kf0) {
  const int lane = threadIdx.x & 31;
  const float* p = base + (size_t)(lane & 15) * ld;
  const int kb = kf0 + ((lane >> 4) << 3);
  v16bf a;
#pragma unroll
  for (int e = 0; e < 16; ++e) {
    int k = kb + (e & 7) + ((e >> 3) << 4);
    a[e] = (__bf16)p[k];
  }
  return a;
}

__device__ __forceinline__ v16bf load_b_bf16(const __bf16* baseT, int ld, int kf0, int kmax) {
  const int lane = threadIdx.x & 31;
  const int kb = kf0 + ((lane >> 4) << 3);
  const __bf16* p = baseT + (size_t)(lane & 15) * ld + kb;
  v16bf b;
#pragma unroll
  for (int e = 0; e < 16; ++e) {
    int k = kb + (e & 7) + ((e >> 3) << 4);
    b[e] = (k < kmax) ? p[k - kb] : (__bf16)0.0f;
  }
  return b;
}

__device__ __forceinline__ unsigned pack_bf16x2(float lo, float hi) {
  union { __bf16 h[2]; unsigned u; } p;
  p.h[0] = (__bf16)lo;
  p.h[1] = (__bf16)hi;
  return p.u;
}

// Generic LDS pointer: hardware uses addr[31:0] as the LDS byte offset.
__device__ __forceinline__ unsigned lds_off(const void* p) {
  return (unsigned)(unsigned long long)p;
}

// ---------------------------------------------------------------------------
// K0: tiled transpose + f32->bf16 convert: src f32 (R, Cc) -> dst bf16 (Cc, R)
// ---------------------------------------------------------------------------
__global__ void transpose_cvt_kernel(const float* __restrict__ src, __bf16* __restrict__ dst,
                                     int R, int Cc, long long srcBatchStride,
                                     long long dstBatchStride) {
  __shared__ float tile[32][33];
  const float* s = src + (size_t)blockIdx.z * srcBatchStride;
  __bf16* d = dst + (size_t)blockIdx.z * dstBatchStride;
  const int c0 = blockIdx.x * 32;
  const int r0 = blockIdx.y * 32;
  const int tx = threadIdx.x, ty = threadIdx.y;
#pragma unroll
  for (int i = 0; i < 32; i += 8)
    tile[ty + i][tx] = s[(size_t)(r0 + ty + i) * Cc + c0 + tx];
  __syncthreads();
#pragma unroll
  for (int i = 0; i < 32; i += 8)
    d[(size_t)(c0 + ty + i) * R + r0 + tx] = (__bf16)tile[tx][ty + i];
}

// ---------------------------------------------------------------------------
// enc (B,77,768) f32 -> encb (B,96,768) bf16, rows l>=77 zeroed
// ---------------------------------------------------------------------------
__global__ void enc_pack_kernel(const float* __restrict__ enc, __bf16* __restrict__ encb) {
  int idx = blockIdx.x * blockDim.x + threadIdx.x;
  if (idx >= B_ * Lp * CD_) return;
  int cd = idx % CD_;
  int l  = (idx / CD_) % Lp;
  int b  = idx / (CD_ * Lp);
  float v = (l < L_) ? enc[((size_t)b * L_ + l) * CD_ + cd] : 0.0f;
  encb[idx] = (__bf16)v;
}

// ---------------------------------------------------------------------------
// K1: K/V projection. encb(b,96,768) x WkT/WvT(640,768) -> per-head packed:
//     kpack(b,h,96,80)  and  vtpack(b,h,80,96)  (bf16, zero-padded rows/cols)
// grid (60, B, 2), block 128: 4 waves x 1 tile each; 6x40 tiles per (b,which)
// ---------------------------------------------------------------------------
__global__ void proj_kv_kernel(const __bf16* __restrict__ encb,
                               const __bf16* __restrict__ WkT,
                               const __bf16* __restrict__ WvT,
                               __bf16* __restrict__ kpack,
                               __bf16* __restrict__ vtpack) {
  const int b = blockIdx.y;
  const bool isV = blockIdx.z != 0;
  const int wave = threadIdx.x >> 5;
  const int lane = threadIdx.x & 31;
  const int tileId = blockIdx.x * 4 + wave;   // 0..239
  const int l0 = (tileId / 40) * 16;
  const int c0 = (tileId % 40) * 16;
  const __bf16* WT = isV ? WvT : WkT;
  const __bf16* abase = encb + ((size_t)b * Lp + l0) * CD_;
  const __bf16* bbase = WT + (size_t)c0 * CD_;
  v8f acc = {};
#pragma unroll 4
  for (int kf = 0; kf < CD_; kf += 32) {
    v16bf a  = load_a_bf16(abase, CD_, kf, CD_);
    v16bf bb = load_b_bf16(bbase, CD_, kf, CD_);
    acc = WMMA_BF16(a, bb, acc);
  }
  const int c = c0 + (lane & 15);
  const int h = c / DH;
  const int d = c - h * DH;
  const int mhi = (lane >> 4) << 3;
  if (!isV) {
    // (l varies per r) -> strided b16 stores
#pragma unroll
    for (int r = 0; r < 8; ++r) {
      const int l = l0 + r + mhi;
      kpack[(((size_t)b * NH + h) * Lp + l) * DH + d] = (__bf16)acc[r];
    }
  } else {
    // vtpack row (d) contiguous in l: 8 consecutive bf16 = one 16B store
    uint4 st;
    st.x = pack_bf16x2(acc[0], acc[1]);
    st.y = pack_bf16x2(acc[2], acc[3]);
    st.z = pack_bf16x2(acc[4], acc[5]);
    st.w = pack_bf16x2(acc[6], acc[7]);
    __bf16* vp = vtpack + (((size_t)b * NH + h) * DH + d) * Lp + l0 + mhi;
    *(uint4*)vp = st;
  }
}

// ---------------------------------------------------------------------------
// K2: Q projection. xb(b,4096,640) x WqT(640,640) -> qb(b,4096,640) bf16
// 16x64 strips: A fragment reused across 4 B tiles. grid (640, B), block 128.
// ---------------------------------------------------------------------------
__global__ void proj_q_kernel(const __bf16* __restrict__ xb,
                              const __bf16* __restrict__ WqT,
                              __bf16* __restrict__ qb) {
  const int b = blockIdx.y;
  const int wave = threadIdx.x >> 5;
  const int lane = threadIdx.x & 31;
  const int sid = blockIdx.x * 4 + wave;  // 0..2559
  const int n0 = (sid / 10) * 16;
  const int c0 = (sid % 10) * 64;
  const __bf16* abase = xb + ((size_t)b * N_ + n0) * C_;
  v8f acc[4] = {{}, {}, {}, {}};
#pragma unroll 2
  for (int kf = 0; kf < C_; kf += 32) {
    v16bf a = load_a_bf16(abase, C_, kf, C_);
#pragma unroll
    for (int j = 0; j < 4; ++j) {
      v16bf bb = load_b_bf16(WqT + (size_t)(c0 + j * 16) * C_, C_, kf, C_);
      acc[j] = WMMA_BF16(a, bb, acc[j]);
    }
  }
  const int mhi = (lane >> 4) << 3;
#pragma unroll
  for (int j = 0; j < 4; ++j) {
    const int c = c0 + j * 16 + (lane & 15);
#pragma unroll
    for (int r = 0; r < 8; ++r)
      qb[((size_t)b * N_ + n0 + r + mhi) * C_ + c] = (__bf16)acc[j][r];
  }
}

// ---------------------------------------------------------------------------
// K3: attention. Per block: one (b,h); K/V staged straight into LDS with
// gfx1250 async global->LDS b128 transfers (ASYNCcnt), no VGPR bounce.
// Each of 4 waves owns a 16-token tile: scores (18 wmma) -> masked softmax
// (lane shuffles within 16-lane C/D row groups) -> probs re-fragment via a
// private LDS tile -> probs x V (15 wmma) -> yattn bf16.
// grid (64, NH, B), block 128.
// ---------------------------------------------------------------------------
__global__ void attention_kernel(const __bf16* __restrict__ qb,
                                 const __bf16* __restrict__ kpack,
                                 const __bf16* __restrict__ vtpack,
                                 __bf16* __restrict__ yattn) {
  __shared__ __align__(16) __bf16 sk[Lp * DH];        // (l, d) 96x80
  __shared__ __align__(16) __bf16 sv[DH * Lp];        // (d, l) 80x96
  __shared__ __align__(16) float  sprobs[4][16 * Lp]; // per-wave 16x96
  const int b = blockIdx.z;
  const int h = blockIdx.y;
  const int wave = threadIdx.x >> 5;
  const int lane = threadIdx.x & 31;

  // Async global->LDS staging: 2 x 960 x 16B transfers across 128 lanes.
  {
    const __bf16* gk = kpack + (size_t)(b * NH + h) * (Lp * DH);
    const __bf16* gv = vtpack + (size_t)(b * NH + h) * (DH * Lp);
    const unsigned lk0 = lds_off(sk);
    const unsigned lv0 = lds_off(sv);
    for (int i = threadIdx.x; i < (Lp * DH) / 8; i += blockDim.x) {
      const unsigned lk = lk0 + i * 16;
      const unsigned lv = lv0 + i * 16;
      const __bf16* gki = gk + i * 8;
      const __bf16* gvi = gv + i * 8;
      asm volatile("global_load_async_to_lds_b128 %0, %1, off"
                   :: "v"(lk), "v"(gki) : "memory");
      asm volatile("global_load_async_to_lds_b128 %0, %1, off"
                   :: "v"(lv), "v"(gvi) : "memory");
    }
    asm volatile("s_wait_asynccnt 0x0" ::: "memory");
  }
  __syncthreads();

  const int n0 = (blockIdx.x * 4 + wave) * 16;
  const __bf16* qbase = qb + ((size_t)b * N_ + n0) * C_ + h * DH;

  // Q A-fragments: K-dim = dh padded 80 -> 96 (zeros; K side equally padded)
  v16bf aq0 = load_a_bf16(qbase, C_, 0,  DH);
  v16bf aq1 = load_a_bf16(qbase, C_, 32, DH);
  v16bf aq2 = load_a_bf16(qbase, C_, 64, DH);

  const float scale = 0.11180339887498949f;  // 1/sqrt(80)
  v8f sacc[6];
#pragma unroll
  for (int lt = 0; lt < 6; ++lt) {
    v8f acc = {};
    const __bf16* kb = sk + lt * 16 * DH;
    acc = WMMA_BF16(aq0, load_b_bf16(kb, DH, 0,  DH), acc);
    acc = WMMA_BF16(aq1, load_b_bf16(kb, DH, 32, DH), acc);
    acc = WMMA_BF16(aq2, load_b_bf16(kb, DH, 64, DH), acc);
    const int l = lt * 16 + (lane & 15);
#pragma unroll
    for (int r = 0; r < 8; ++r)
      acc[r] = (l < L_) ? acc[r] * scale : -1e30f;
    sacc[lt] = acc;
  }

  // Row-wise softmax: row m = r + 8*(lane>>4) lives in one 16-lane group.
  const int mhi = (lane >> 4) << 3;
  float* sp = sprobs[wave];
#pragma unroll
  for (int r = 0; r < 8; ++r) {
    float mx = sacc[0][r];
#pragma unroll
    for (int lt = 1; lt < 6; ++lt) mx = fmaxf(mx, sacc[lt][r]);
#pragma unroll
    for (int off = 8; off >= 1; off >>= 1) mx = fmaxf(mx, __shfl_xor(mx, off, 32));
    float sum = 0.0f;
#pragma unroll
    for (int lt = 0; lt < 6; ++lt) {
      float e = __expf(sacc[lt][r] - mx);  // masked cols -> exp(-huge) = 0
      sacc[lt][r] = e;
      sum += e;
    }
#pragma unroll
    for (int off = 8; off >= 1; off >>= 1) sum += __shfl_xor(sum, off, 32);
    const float inv = 1.0f / sum;
#pragma unroll
    for (int lt = 0; lt < 6; ++lt)
      sp[(r + mhi) * Lp + lt * 16 + (lane & 15)] = sacc[lt][r] * inv;
  }
  // Same-wave LDS RAW; in-order DS + compiler s_wait_dscnt covers it.

  v16bf ap0 = load_a_f32(sp, Lp, 0);
  v16bf ap1 = load_a_f32(sp, Lp, 32);
  v16bf ap2 = load_a_f32(sp, Lp, 64);

#pragma unroll
  for (int dt = 0; dt < 5; ++dt) {
    v8f acc = {};
    const __bf16* vb = sv + dt * 16 * Lp;
    acc = WMMA_BF16(ap0, load_b_bf16(vb, Lp, 0,  Lp), acc);
    acc = WMMA_BF16(ap1, load_b_bf16(vb, Lp, 32, Lp), acc);
    acc = WMMA_BF16(ap2, load_b_bf16(vb, Lp, 64, Lp), acc);
    const int d = dt * 16 + (lane & 15);
#pragma unroll
    for (int r = 0; r < 8; ++r)
      yattn[((size_t)b * N_ + n0 + r + mhi) * C_ + h * DH + d] = (__bf16)acc[r];
  }
}

// ---------------------------------------------------------------------------
// K4: output projection + bias, fp32 store transposed to (B, C, N)
// ---------------------------------------------------------------------------
__global__ void proj_out_kernel(const __bf16* __restrict__ yattn,
                                const __bf16* __restrict__ WoT,
                                const float* __restrict__ bo,
                                float* __restrict__ out) {
  const int b = blockIdx.y;
  const int wave = threadIdx.x >> 5;
  const int lane = threadIdx.x & 31;
  const int sid = blockIdx.x * 4 + wave;
  const int n0 = (sid / 10) * 16;
  const int c0 = (sid % 10) * 64;
  const __bf16* abase = yattn + ((size_t)b * N_ + n0) * C_;
  v8f acc[4] = {{}, {}, {}, {}};
#pragma unroll 2
  for (int kf = 0; kf < C_; kf += 32) {
    v16bf a = load_a_bf16(abase, C_, kf, C_);
#pragma unroll
    for (int j = 0; j < 4; ++j) {
      v16bf bb = load_b_bf16(WoT + (size_t)(c0 + j * 16) * C_, C_, kf, C_);
      acc[j] = WMMA_BF16(a, bb, acc[j]);
    }
  }
  const int mhi = (lane >> 4) << 3;
#pragma unroll
  for (int j = 0; j < 4; ++j) {
    const int c = c0 + j * 16 + (lane & 15);
    const float bias = bo[c];
    float* obase = out + ((size_t)b * C_ + c) * N_ + n0 + mhi;
    float4 v0, v1;  // 8 consecutive n per lane -> two b128 stores
    v0.x = acc[j][0] + bias; v0.y = acc[j][1] + bias;
    v0.z = acc[j][2] + bias; v0.w = acc[j][3] + bias;
    v1.x = acc[j][4] + bias; v1.y = acc[j][5] + bias;
    v1.z = acc[j][6] + bias; v1.w = acc[j][7] + bias;
    *(float4*)obase = v0;
    *(float4*)(obase + 4) = v1;
  }
}

// saliency = mean over heads then over L of softmax rows == exactly 1/77
__global__ void saliency_kernel(float* __restrict__ s, int n) {
  int i = blockIdx.x * blockDim.x + threadIdx.x;
  if (i < n) s[i] = 1.0f / 77.0f;
}

extern "C" void kernel_launch(void* const* d_in, const int* in_sizes, int n_in,
                              void* d_out, int out_size, void* d_ws, size_t ws_size,
                              hipStream_t stream) {
  const float* hidden = (const float*)d_in[0];  // (8,640,64,64)
  const float* enc    = (const float*)d_in[1];  // (8,77,768)
  const float* Wq     = (const float*)d_in[2];  // (640,640)
  const float* Wk     = (const float*)d_in[3];  // (768,640)
  const float* Wv     = (const float*)d_in[4];  // (768,640)
  const float* Wo     = (const float*)d_in[5];  // (640,640)
  const float* bo     = (const float*)d_in[6];  // (640,)
  float* out = (float*)d_out;
  float* saliency = out + (size_t)B_ * C_ * N_;

  char* ws = (char*)d_ws;
  size_t off = 0;
  auto alloc = [&](size_t bytes) -> char* {
    char* p = ws + off;
    off += (bytes + 255) & ~(size_t)255;
    return p;
  };
  __bf16* xb     = (__bf16*)alloc((size_t)B_ * N_ * C_ * 2);   // 41.9 MB
  __bf16* qb     = (__bf16*)alloc((size_t)B_ * N_ * C_ * 2);   // 41.9 MB
  __bf16* WqT    = (__bf16*)alloc((size_t)C_ * C_ * 2);
  __bf16* WkT    = (__bf16*)alloc((size_t)C_ * CD_ * 2);
  __bf16* WvT    = (__bf16*)alloc((size_t)C_ * CD_ * 2);
  __bf16* WoT    = (__bf16*)alloc((size_t)C_ * C_ * 2);
  __bf16* encb   = (__bf16*)alloc((size_t)B_ * Lp * CD_ * 2);
  __bf16* kpack  = (__bf16*)alloc((size_t)B_ * NH * Lp * DH * 2);
  __bf16* vtpack = (__bf16*)alloc((size_t)B_ * NH * DH * Lp * 2);
  __bf16* yattn  = xb;  // xb dead after proj_q; alias to save ~42 MB

  const dim3 tb(32, 8);
  // K0: transposes + bf16 conversion
  transpose_cvt_kernel<<<dim3(N_ / 32, C_ / 32, B_), tb, 0, stream>>>(
      hidden, xb, C_, N_, (long long)C_ * N_, (long long)N_ * C_);
  transpose_cvt_kernel<<<dim3(C_ / 32, C_ / 32, 1), tb, 0, stream>>>(
      Wq, WqT, C_, C_, 0, 0);
  transpose_cvt_kernel<<<dim3(C_ / 32, CD_ / 32, 1), tb, 0, stream>>>(
      Wk, WkT, CD_, C_, 0, 0);
  transpose_cvt_kernel<<<dim3(C_ / 32, CD_ / 32, 1), tb, 0, stream>>>(
      Wv, WvT, CD_, C_, 0, 0);
  transpose_cvt_kernel<<<dim3(C_ / 32, C_ / 32, 1), tb, 0, stream>>>(
      Wo, WoT, C_, C_, 0, 0);
  enc_pack_kernel<<<(B_ * Lp * CD_ + 255) / 256, 256, 0, stream>>>(enc, encb);

  // K1: K/V projections into per-head packed layouts
  proj_kv_kernel<<<dim3(60, B_, 2), 128, 0, stream>>>(encb, WkT, WvT, kpack, vtpack);
  // K2: Q projection
  proj_q_kernel<<<dim3(640, B_), 128, 0, stream>>>(xb, WqT, qb);
  // K3: attention (async global->LDS staging + WMMA)
  attention_kernel<<<dim3(64, NH, B_), 128, 0, stream>>>(qb, kpack, vtpack, yattn);
  // K4: output projection (+bias, transposed fp32 b128 stores)
  proj_out_kernel<<<dim3(640, B_), 128, 0, stream>>>(yattn, WoT, bo, out);
  // K5: saliency fill
  saliency_kernel<<<(B_ * 64 * 64 + 255) / 256, 256, 0, stream>>>(saliency, B_ * 64 * 64);
}
